// SRNN_88441966559369
// MI455X (gfx1250) — matrix operand order repaired and checked by reading, not compiled, not run on hardware
//
#include <hip/hip_runtime.h>
#include <hip/hip_bf16.h>

typedef __attribute__((ext_vector_type(16))) __bf16 v16bf;
typedef __attribute__((ext_vector_type(8)))  float  v8f;
typedef unsigned short u16;

#define N_B    128
#define INP_D  256
#define N_T    1000
#define N_REC  512
#define OUT_D  32
#define ALPHA_C 0.9512294245007140f   /* exp(-0.001/0.02) */
#define KAPPA_C 0.9512294245007140f
#define THR_C   0.6f

static __device__ __forceinline__ u16 f2bf(float v) {
    __hip_bfloat16 h = __float2bfloat16(v);
    u16 u; __builtin_memcpy(&u, &h, 2); return u;
}

// ---- A-fragment (16-bit A 16x32): lane holds M=lane&15; lanes0-15 K=base..base+7
// in VGPR0-3 and K=base+16..base+23 in VGPR4-7; lanes16-31 shifted by 8.
static __device__ __forceinline__ v16bf frag_gap(const u16* p) {
    union { uint4 q[2]; v16bf v; } u;
    u.q[0] = *(const uint4*)(p);        // 8 halves
    u.q[1] = *(const uint4*)(p + 16);   // 8 halves, +16 halfword gap
    return u.v;
}
// ---- contiguous loader: 16 consecutive K values from p (unused; B fragments use frag_gap).
static __device__ __forceinline__ v16bf frag_cont(const u16* p) {
    union { uint4 q[2]; v16bf v; } u;
    u.q[0] = *(const uint4*)(p);
    u.q[1] = *(const uint4*)(p + 8);
    return u.v;
}

// ===================== prep: weights -> bf16 (w_rec diagonal zeroed) ==========
__global__ void prep_weights(const float* __restrict__ w_inp,
                             const float* __restrict__ w_rec,
                             const float* __restrict__ w_out,
                             u16* __restrict__ wrec_b,
                             u16* __restrict__ winp_b,
                             u16* __restrict__ wout_b) {
    int idx = blockIdx.x * 256 + threadIdx.x;
    if (idx < N_REC * N_REC) {
        int r = idx >> 9, k = idx & (N_REC - 1);
        float v = (r == k) ? 0.f : w_rec[idx];
        wrec_b[idx] = f2bf(v);
    }
    if (idx < N_REC * INP_D) winp_b[idx] = f2bf(w_inp[idx]);
    if (idx < OUT_D * N_REC) wout_b[idx] = f2bf(w_out[idx]);
}

// ===================== prep: x[b][i][t] fp32 -> xt[t][b][i] bf16 ==============
__global__ void transpose_x(const float* __restrict__ x, u16* __restrict__ xt) {
    __shared__ float tile[32][33];
    int b  = blockIdx.z;
    int i0 = blockIdx.y * 32;
    int t0 = blockIdx.x * 32;
    int lt = threadIdx.x & 31;
    int lr = threadIdx.x >> 5;               // 8 rows of 32 lanes
    #pragma unroll
    for (int rr = lr; rr < 32; rr += 8) {
        int t = t0 + lt;
        float v = (t < N_T) ? x[((size_t)b * INP_D + (i0 + rr)) * N_T + t] : 0.f;
        tile[rr][lt] = v;
    }
    __syncthreads();
    #pragma unroll
    for (int rr = lr; rr < 32; rr += 8) {
        int t = t0 + rr;
        if (t < N_T)
            xt[((size_t)t * N_B + b) * INP_D + (i0 + lt)] = f2bf(tile[lt][rr]);
    }
}

// ===================== main persistent recurrence kernel ======================
// grid = 8 blocks (one per 16-batch tile), 256 threads = 8 waves.
// Wave w owns N-tiles 4w..4w+3 (neurons 64w..64w+63). vo handled by waves 0,1.
__launch_bounds__(256, 1)
__global__ void srnn_main(const u16* __restrict__ xt,
                          const u16* __restrict__ wrec,
                          const u16* __restrict__ winp,
                          const u16* __restrict__ wout,
                          float* __restrict__ out) {
    __shared__ __align__(16) u16 zbuf[2][16 * N_REC];   // 2 x 16KB spike buffers

    const int tid  = threadIdx.x;
    const int lane = tid & 31;
    const int wv   = tid >> 5;       // wave 0..7
    const int m16  = lane & 15;
    const int hi   = lane >> 4;      // 0/1 half-wave
    const int bm   = blockIdx.x * 16;

    // z_0 = 0
    for (int i = tid; i < 2 * 16 * N_REC; i += 256) ((u16*)zbuf)[i] = 0;
    // vo row t=0 is zero in the reference output
    for (int i = tid; i < 16 * OUT_D; i += 256) {
        int r = i >> 5, o = i & 31;
        out[(size_t)(bm + r) * OUT_D + o] = 0.f;
    }

    v8f vst[4];                       // membrane potential v, fp32, persistent
    v8f vo = {0.f,0.f,0.f,0.f,0.f,0.f,0.f,0.f};
    #pragma unroll
    for (int nt = 0; nt < 4; ++nt) vst[nt] = vo;

    __syncthreads();

    int cur = 0;
    for (int t = 0; t < N_T - 1; ++t) {
        // ---- v_{t+1} = ALPHA*v_t - THR*z_t  (z_t recomputed from v_t) --------
        v8f c[4];
        #pragma unroll
        for (int nt = 0; nt < 4; ++nt) {
            #pragma unroll
            for (int j = 0; j < 8; ++j) {
                float v = vst[nt][j];
                c[nt][j] = ALPHA_C * v - ((v > THR_C) ? THR_C : 0.f);
            }
        }
        // ---- + x_t @ W_inp^T  (K = 256, bf16 WMMA) ---------------------------
        {
            const u16* arow = xt + ((size_t)t * N_B + bm + m16) * INP_D;
            #pragma unroll 4
            for (int kt = 0; kt < INP_D / 32; ++kt) {
                v16bf a = frag_gap(arow + kt * 32 + hi * 8);
                #pragma unroll
                for (int nt = 0; nt < 4; ++nt) {
                    int n = (wv * 4 + nt) * 16 + m16;
                    v16bf b = frag_gap(winp + (size_t)n * INP_D + kt * 32 + hi * 8);
                    c[nt] = __builtin_amdgcn_wmma_f32_16x16x32_bf16(
                        false, a, false, b, (short)0, c[nt], false, false);
                }
            }
        }
        // ---- + z_t @ W_rec^T  (K = 512, A from LDS) --------------------------
        {
            const u16* zrow = &zbuf[cur][m16 * N_REC];
            #pragma unroll 4
            for (int kt = 0; kt < N_REC / 32; ++kt) {
                v16bf a = frag_gap(zrow + kt * 32 + hi * 8);
                #pragma unroll
                for (int nt = 0; nt < 4; ++nt) {
                    int n = (wv * 4 + nt) * 16 + m16;
                    v16bf b = frag_gap(wrec + (size_t)n * N_REC + kt * 32 + hi * 8);
                    c[nt] = __builtin_amdgcn_wmma_f32_16x16x32_bf16(
                        false, a, false, b, (short)0, c[nt], false, false);
                }
            }
        }
        // ---- z_{t+1} -> other LDS buffer; commit v ---------------------------
        {
            u16* zo = &zbuf[cur ^ 1][0];
            #pragma unroll
            for (int nt = 0; nt < 4; ++nt) {
                int n = (wv * 4 + nt) * 16 + m16;
                #pragma unroll
                for (int j = 0; j < 8; ++j) {
                    int mrow = j + hi * 8;
                    zo[mrow * N_REC + n] = (c[nt][j] > THR_C) ? (u16)0x3F80 : (u16)0;
                }
                vst[nt] = c[nt];
            }
        }
        __syncthreads();
        // ---- vo_{t+1} = KAPPA*vo_t + z_{t+1} @ W_out^T  (waves 0,1) ----------
        if (wv < 2) {
            v8f co;
            #pragma unroll
            for (int j = 0; j < 8; ++j) co[j] = KAPPA_C * vo[j];
            const u16* zrow = &zbuf[cur ^ 1][m16 * N_REC];
            int o = wv * 16 + m16;
            #pragma unroll 2
            for (int kt = 0; kt < N_REC / 32; ++kt) {
                v16bf a = frag_gap(zrow + kt * 32 + hi * 8);
                v16bf b = frag_gap(wout + (size_t)o * N_REC + kt * 32 + hi * 8);
                co = __builtin_amdgcn_wmma_f32_16x16x32_bf16(
                    false, a, false, b, (short)0, co, false, false);
            }
            vo = co;
            #pragma unroll
            for (int j = 0; j < 8; ++j) {
                int mrow = j + hi * 8;
                out[((size_t)(t + 1) * N_B + bm + mrow) * OUT_D + o] = co[j];
            }
        }
        cur ^= 1;
    }
}

// ===================== host launcher =========================================
extern "C" void kernel_launch(void* const* d_in, const int* in_sizes, int n_in,
                              void* d_out, int out_size, void* d_ws, size_t ws_size,
                              hipStream_t stream) {
    const float* x     = (const float*)d_in[0];
    const float* w_inp = (const float*)d_in[1];
    const float* w_rec = (const float*)d_in[2];
    const float* w_out = (const float*)d_in[3];
    float* out = (float*)d_out;

    char* ws = (char*)d_ws;
    u16* wrec_b = (u16*)(ws);                      // 512*512*2 = 512 KB
    u16* winp_b = (u16*)(ws + (512 * 1024));       // 512*256*2 = 256 KB
    u16* wout_b = (u16*)(ws + (768 * 1024));       //  32*512*2 =  32 KB
    u16* xt     = (u16*)(ws + (1024 * 1024));      // 1000*128*256*2 = 62.5 MB

    prep_weights<<<(N_REC * N_REC + 255) / 256, 256, 0, stream>>>(
        w_inp, w_rec, w_out, wrec_b, winp_b, wout_b);

    dim3 tg((N_T + 31) / 32, INP_D / 32, N_B);
    transpose_x<<<tg, 256, 0, stream>>>(x, xt);

    srnn_main<<<N_B / 16, 256, 0, stream>>>(xt, wrec_b, winp_b, wout_b, out);
}
